// FeedForward_17025250361617
// MI455X (gfx1250) — matrix operand order repaired and checked by reading, not compiled, not run on hardware
//
#include <hip/hip_runtime.h>
#include <math.h>

#define D_DIM 1024
#define H_DIM 4096
#define E_NUM 8
#define NPAIR 8192        // B*S*K
#define SLAB_STRIDE 129   // f32 slab row stride in dwords (128 data + 1 TDM pad dword)

typedef __attribute__((ext_vector_type(16))) __bf16 v16bf;
typedef __attribute__((ext_vector_type(8)))  __bf16 v8bf;
typedef __attribute__((ext_vector_type(8)))  float  v8f;
typedef __attribute__((ext_vector_type(4)))  unsigned int u32x4;
typedef __attribute__((ext_vector_type(8)))  int i32x8;
typedef __attribute__((ext_vector_type(4)))  int i32x4;

union AB { v16bf v; v8bf h[2]; };

#if defined(__HIP_DEVICE_COMPILE__) && defined(__gfx1250__)
#  if __has_builtin(__builtin_amdgcn_tensor_load_to_lds) && __has_builtin(__builtin_amdgcn_s_wait_tensorcnt)
#    define USE_TDM 1
#  endif
#endif
#ifndef USE_TDM
#  define USE_TDM 0
#endif

__device__ __forceinline__ float gelu_tanh(float x) {
    float x3 = x * x * x;
    float t  = tanhf(0.7978845608028654f * (x + 0.044715f * x3));
    return 0.5f * x * (1.0f + t);
}

#if USE_TDM
__device__ __forceinline__ unsigned rf(unsigned v) {
    return (unsigned)__builtin_amdgcn_readfirstlane((int)v);
}

// 2D tile load: tile 128(rows, k-dim) x 128(cols) f32, row-padded by 1 dword in LDS.
// D# per CDNA5 ISA: group0 = {flags, lds_addr, global_addr, type=2},
// group1 = {data_size=4B | pad(128dw interval, 1dw amount), tensor dims, tile dims, stride}.
__device__ __forceinline__ void tdm_load_slab(unsigned lds_off, const float* g,
                                              unsigned dim0, unsigned dim1, unsigned stride)
{
    unsigned long long ga = (unsigned long long)(size_t)g;
    u32x4 g0;
    g0[0] = rf(1u);                                               // count=1
    g0[1] = rf(lds_off);                                          // lds_addr
    g0[2] = rf((unsigned)ga);                                     // global_addr[31:0]
    g0[3] = rf(((unsigned)(ga >> 32) & 0x01FFFFFFu) | (2u << 30));// addr[56:32] | type=2
    i32x8 g1;
    g1[0] = (int)rf((2u << 16) | (1u << 20) | (6u << 22));        // 4B elems, pad_en, 128dw interval, 1dw pad
    g1[1] = (int)rf((dim0 & 0xFFFFu) << 16);                      // tensor_dim0[15:0]
    g1[2] = (int)rf(((dim0 >> 16) & 0xFFFFu) | ((dim1 & 0xFFFFu) << 16));
    g1[3] = (int)rf(((dim1 >> 16) & 0xFFFFu) | (128u << 16));     // tile_dim0 = 128
    g1[4] = (int)rf(128u);                                        // tile_dim1 = 128
    g1[5] = (int)rf(stride);                                      // tensor_dim0_stride (elems)
    g1[6] = 0; g1[7] = 0;
    i32x4 z4 = {0, 0, 0, 0};
    i32x8 z8 = {0, 0, 0, 0, 0, 0, 0, 0};
    __builtin_amdgcn_tensor_load_to_lds(g0, g1, z4, z4, z8, 0);
}

// Gather-mode load: 16 full rows of x (1024 f32 each) selected by 16-bit row indices.
__device__ __forceinline__ void tdm_gather_rows(unsigned lds_off, const float* g,
                                                i32x4 g2, i32x4 g3)
{
    unsigned long long ga = (unsigned long long)(size_t)g;
    u32x4 g0;
    g0[0] = rf(1u | (1u << 31));                                  // count=1 | gather_mode (16-bit idx)
    g0[1] = rf(lds_off);
    g0[2] = rf((unsigned)ga);
    g0[3] = rf(((unsigned)(ga >> 32) & 0x01FFFFFFu) | (2u << 30));
    i32x8 g1;
    g1[0] = (int)rf(2u << 16);                                    // 4B elems, no pad
    g1[1] = (int)rf((unsigned)D_DIM << 16);                       // tensor_dim0 = 1024
    g1[2] = (int)rf(((unsigned)(D_DIM >> 16)) | (4096u << 16));   // tensor_dim1 = 4096 rows
    g1[3] = (int)rf((unsigned)D_DIM << 16);                       // tile_dim0 = 1024
    g1[4] = (int)rf(16u);                                         // tile_dim1 = 16 valid indices
    g1[5] = (int)rf((unsigned)D_DIM);                             // stride = 1024
    g1[6] = 0; g1[7] = 0;
    i32x8 z8 = {0, 0, 0, 0, 0, 0, 0, 0};
    __builtin_amdgcn_tensor_load_to_lds(g0, g1, g2, g3, z8, 0);
}
#endif // USE_TDM

// ---------------- routing / init kernels ----------------

__global__ void zero_out_kernel(float* out, int n) {
    int i = blockIdx.x * blockDim.x + threadIdx.x;
    if (i < n) out[i] = 0.0f;
}

__global__ void zero_meta_kernel(int* meta) {
    int i = threadIdx.x;
    if (i < 16) meta[i] = 0;
}

__global__ void count_kernel(const int* __restrict__ idx, int* counts) {
    int i = blockIdx.x * blockDim.x + threadIdx.x;
    if (i < NPAIR) atomicAdd(&counts[idx[i]], 1);
}

__global__ void offsets_kernel(const int* __restrict__ counts, int* offsets) {
    if (threadIdx.x == 0) {
        int acc = 0;
        for (int e = 0; e < E_NUM; ++e) { offsets[e] = acc; acc += counts[e]; }
    }
}

__global__ void scatter_kernel(const int* __restrict__ idx, int* cursor,
                               const int* __restrict__ offsets, int* list) {
    int i = blockIdx.x * blockDim.x + threadIdx.x;
    if (i < NPAIR) {
        int e = idx[i];
        int p = atomicAdd(&cursor[e], 1);
        list[offsets[e] + p] = i;     // i = token*2 + k
    }
}

// ---------------- fused grouped-GEMM tile kernel ----------------

__global__ __launch_bounds__(256)
void moe_tile_kernel(const float* __restrict__ x,    const float* __restrict__ wts,
                     const float* __restrict__ keys, const float* __restrict__ values,
                     float* __restrict__ out,
                     const int* __restrict__ counts, const int* __restrict__ offsets,
                     const int* __restrict__ list)
{
    // 2 x 66 KB double-buffered f32 weight slabs (TDM destination, row-padded),
    // buffer 0 also reused once for the gathered x rows (16*1024 f32 <= 16512*1).
    __shared__ __align__(16) float  sf[2][128 * SLAB_STRIDE];
    __shared__ __align__(32) __bf16 xs[16 * D_DIM];   // 32 KB  x tile (bf16)
    __shared__ __align__(32) __bf16 hs[16 * H_DIM];   // 128 KB hidden tile (bf16)
    __shared__ int   tok[16];
    __shared__ float wrow[16];

    const int tid  = threadIdx.x;
    const int lane = tid & 31;
    const int wave = tid >> 5;          // 8 wave32 waves
    const int half = lane >> 4;
    const int l16  = lane & 15;

    // map blockIdx.x -> (expert, local tile index)
    int e = -1, local = blockIdx.x;
    for (int i = 0; i < E_NUM; ++i) {
        int nt = (counts[i] + 15) >> 4;
        if (local < nt) { e = i; break; }
        local -= nt;
    }
    if (e < 0) return;
    const int cnt  = counts[e];
    const int base = offsets[e];

    if (tid < 16) {
        int r = (local << 4) + tid;
        if (r < cnt) { int ent = list[base + r]; tok[tid] = ent >> 1; wrow[tid] = wts[ent]; }
        else         { tok[tid] = 0;             wrow[tid] = 0.0f; }
    }
    __syncthreads();

    const float* keysE = keys   + (size_t)e * D_DIM * H_DIM;   // [d][h], h contiguous
    const float* valsE = values + (size_t)e * H_DIM * D_DIM;   // [h][d], d contiguous

    // ---- stage the 16 token rows of x into sf[0] ----
#if USE_TDM
    if (wave == 0) {
        i32x4 g2, g3;
        for (int j = 0; j < 4; ++j)
            g2[j] = (int)rf((unsigned)tok[2*j]   | ((unsigned)tok[2*j+1]   << 16));
        for (int j = 0; j < 4; ++j)
            g3[j] = (int)rf((unsigned)tok[8+2*j] | ((unsigned)tok[9+2*j]   << 16));
        tdm_gather_rows((unsigned)(size_t)(void*)&sf[0][0], x, g2, g3);
        __builtin_amdgcn_s_wait_tensorcnt(0);
    }
    __syncthreads();
#else
    for (int i = tid; i < 16 * 256; i += 256) {
        int row = i >> 8, c4 = (i & 255) << 2;
        const float4 v = *(const float4*)(x + (size_t)tok[row] * D_DIM + c4);
        *(float4*)&sf[0][row * D_DIM + c4] = v;
    }
    __syncthreads();
#endif
    // convert x rows f32 -> bf16 (coalesced b128 LDS reads, b64 stores)
    for (int i = tid; i < 4096; i += 256) {
        const float4 v = *(const float4*)&sf[0][i << 2];
        __bf16* p = &xs[i << 2];
        p[0] = (__bf16)v.x; p[1] = (__bf16)v.y; p[2] = (__bf16)v.z; p[3] = (__bf16)v.w;
    }

#if USE_TDM
    const unsigned sfoff[2] = { (unsigned)(size_t)(void*)&sf[0][0],
                                (unsigned)(size_t)(void*)&sf[1][0] };
#endif

    // ---------------- GEMM1: hidden[16,4096] = gelu(X @ keys[e]) ----------------
    for (int hb = 0; hb < H_DIM / 128; ++hb) {
        v8f acc = {};
        int cur = 0;
#if USE_TDM
        __syncthreads();                 // all readers of previous slabs (and xs convert) done
        if (wave == 0)
            tdm_load_slab(sfoff[0], keysE + (size_t)0 * H_DIM + hb * 128, H_DIM, D_DIM, H_DIM);
#endif
        for (int kc = 0; kc < D_DIM / 128; ++kc) {
            __syncthreads();
#if USE_TDM
            if (wave == 0) {
                if (kc + 1 < D_DIM / 128) {   // prefetch next slab via TDM, overlap with WMMA
                    tdm_load_slab(sfoff[cur ^ 1],
                                  keysE + (size_t)((kc + 1) * 128) * H_DIM + hb * 128,
                                  H_DIM, D_DIM, H_DIM);
                    __builtin_amdgcn_s_wait_tensorcnt(1);
                } else {
                    __builtin_amdgcn_s_wait_tensorcnt(0);
                }
            }
#else
            for (int i = tid; i < 4096; i += 256) {
                int dl = i >> 5, h4 = (i & 31) << 2;
                const float4 v = *(const float4*)(keysE + (size_t)(kc * 128 + dl) * H_DIM
                                                        + hb * 128 + h4);
                float* p = &sf[0][dl * SLAB_STRIDE + h4];
                p[0] = v.x; p[1] = v.y; p[2] = v.z; p[3] = v.w;
            }
#endif
            __syncthreads();
            const float* slab = &sf[cur][0];
            const int kbase = kc * 128;
#pragma unroll
            for (int ks = 0; ks < 4; ++ks) {
                AB a;
                // A 16x32 bf16: lanes 0-15 K={0..15}, lanes 16-31 K={16..31}
                a.h[0] = *(const v8bf*)&xs[l16 * D_DIM + kbase + ks * 32 + half * 16];
                a.h[1] = *(const v8bf*)&xs[l16 * D_DIM + kbase + ks * 32 + half * 16 + 8];
                // B 32x16: lane n = l16, k = half*16 + i; stride-129 rows -> conflict-free
                const float* bp = &slab[(size_t)(ks * 32 + half * 16) * SLAB_STRIDE
                                        + wave * 16 + l16];
                v16bf bv;
#pragma unroll
                for (int i = 0; i < 16; ++i) bv[i] = (__bf16)bp[i * SLAB_STRIDE];
                acc = __builtin_amdgcn_wmma_f32_16x16x32_bf16(
                          false, a.v, false, bv, (short)0, acc, false, false);
            }
#if USE_TDM
            cur ^= 1;
#endif
        }
#pragma unroll
        for (int r = 0; r < 8; ++r) {     // D layout: m = r + 8*half, n = l16
            int m = r + (half << 3);
            hs[m * H_DIM + hb * 128 + wave * 16 + l16] = (__bf16)gelu_tanh(acc[r]);
        }
    }

    // ---------------- GEMM2: out16[16,1024] = hidden @ values[e] ----------------
    for (int db = 0; db < D_DIM / 128; ++db) {
        v8f acc = {};
        int cur = 0;
#if USE_TDM
        __syncthreads();
        if (wave == 0)
            tdm_load_slab(sfoff[0], valsE + (size_t)0 * D_DIM + db * 128, D_DIM, H_DIM, D_DIM);
#endif
        for (int kc = 0; kc < H_DIM / 128; ++kc) {
            __syncthreads();
#if USE_TDM
            if (wave == 0) {
                if (kc + 1 < H_DIM / 128) {
                    tdm_load_slab(sfoff[cur ^ 1],
                                  valsE + (size_t)((kc + 1) * 128) * D_DIM + db * 128,
                                  D_DIM, H_DIM, D_DIM);
                    __builtin_amdgcn_s_wait_tensorcnt(1);
                } else {
                    __builtin_amdgcn_s_wait_tensorcnt(0);
                }
            }
#else
            for (int i = tid; i < 4096; i += 256) {
                int hl = i >> 5, d4 = (i & 31) << 2;
                const float4 v = *(const float4*)(valsE + (size_t)(kc * 128 + hl) * D_DIM
                                                        + db * 128 + d4);
                float* p = &sf[0][hl * SLAB_STRIDE + d4];
                p[0] = v.x; p[1] = v.y; p[2] = v.z; p[3] = v.w;
            }
#endif
            __syncthreads();
            const float* slab = &sf[cur][0];
            const int kbase = kc * 128;
#pragma unroll
            for (int ks = 0; ks < 4; ++ks) {
                AB a;
                a.h[0] = *(const v8bf*)&hs[l16 * H_DIM + kbase + ks * 32 + half * 16];
                a.h[1] = *(const v8bf*)&hs[l16 * H_DIM + kbase + ks * 32 + half * 16 + 8];
                const float* bp = &slab[(size_t)(ks * 32 + half * 16) * SLAB_STRIDE
                                        + wave * 16 + l16];
                v16bf bv;
#pragma unroll
                for (int i = 0; i < 16; ++i) bv[i] = (__bf16)bp[i * SLAB_STRIDE];
                acc = __builtin_amdgcn_wmma_f32_16x16x32_bf16(
                          false, a.v, false, bv, (short)0, acc, false, false);
            }
#if USE_TDM
            cur ^= 1;
#endif
        }
#pragma unroll
        for (int r = 0; r < 8; ++r) {
            int m = r + (half << 3);
            atomicAdd(&out[(size_t)tok[m] * D_DIM + db * 128 + wave * 16 + l16],
                      wrow[m] * acc[r]);
        }
    }
}

// ---------------- host-side launch ----------------

extern "C" void kernel_launch(void* const* d_in, const int* in_sizes, int n_in,
                              void* d_out, int out_size, void* d_ws, size_t ws_size,
                              hipStream_t stream)
{
    const float* x    = (const float*)d_in[0];
    const int*   idx  = (const int*)d_in[1];
    const float* wts  = (const float*)d_in[2];
    const float* keys = (const float*)d_in[3];
    const float* vals = (const float*)d_in[4];
    float* out = (float*)d_out;

    int* meta    = (int*)d_ws;       // [0..7] counts, [8..15] cursor, [16..23] offsets
    int* counts  = meta;
    int* cursor  = meta + 8;
    int* offsets = meta + 16;
    int* list    = meta + 24;        // NPAIR ints

    zero_out_kernel<<<(out_size + 255) / 256, 256, 0, stream>>>(out, out_size);
    zero_meta_kernel<<<1, 64, 0, stream>>>(meta);
    count_kernel<<<(NPAIR + 255) / 256, 256, 0, stream>>>(idx, counts);
    offsets_kernel<<<1, 32, 0, stream>>>(counts, offsets);
    scatter_kernel<<<(NPAIR + 255) / 256, 256, 0, stream>>>(idx, cursor, offsets, list);

    moe_tile_kernel<<<520, 256, 0, stream>>>(x, wts, keys, vals, out,
                                             counts, offsets, list);
}
